// tfESN_60146722013580
// MI455X (gfx1250) — compile-verified, matrix operand order checked by reading, not run on hardware
//
#include <hip/hip_runtime.h>
#include <stdint.h>

// ---------------------------------------------------------------------------
// ESN step: out = 0.1*x + 0.9*tanh(passed@W_in^T + x@W_res^T + toUse^T@W_out^T + B_in)
//
// Fast path (needs ~42MB ws):
//   1) esn_pack: convert [passed|x|toUse^T] and [W_in|W_res|W_out] to bf16,
//      in WMMA-fragment tile order (8KB per 128x32 tile).
//   2) esn_wmma_async: 128x256 block tile, 64x64 wave tile (4x4 WMMA tiles),
//      double-buffered GLOBAL_LOAD_ASYNC_TO_LDS_B128 pipeline feeding
//      v_wmma_f32_16x16x32_bf16, fused tanh/blend epilogue.
// Fallback (256KB ws): inline fp32->bf16 convert GEMM.
// ---------------------------------------------------------------------------

typedef __attribute__((ext_vector_type(16))) __bf16 v16bf;
typedef __attribute__((ext_vector_type(8)))  __bf16 v8bf;
typedef __attribute__((ext_vector_type(8)))  float  v8f;

#define N_RES 4096
#define BATCH 1024
#define N_IN  128
#define N_OUT 64
#define KTOT   (N_IN + N_RES + N_OUT)  // 4288
#define KSTEPS (KTOT / 32)             // 134
#define ABLK   (BATCH / 128)           // 8
#define WBLK   (N_RES / 128)           // 32
#define TILE_HALVES 4096               // 128 rows x 32 k, bf16 (8KB)
#define BUF_HALVES  12288              // A tile (4096) + two B tiles (8192)

// ---------------------------------------------------------------------------
// CDNA5 async copy helpers (ASYNCcnt-tracked; ISA 15.18.3 opcode 98)
// ---------------------------------------------------------------------------
__device__ __forceinline__ void async_b128(uint32_t lds_off, const void* gaddr) {
  // vdst = LDS byte offset VGPR, vaddr = 64-bit global address VGPR pair
  asm volatile("global_load_async_to_lds_b128 %0, %1, off"
               :: "v"(lds_off), "v"(gaddr) : "memory");
}
__device__ __forceinline__ void wait_async_le6() {
#if __has_builtin(__builtin_amdgcn_s_wait_asynccnt)
  __builtin_amdgcn_s_wait_asynccnt(6);
#else
  asm volatile("s_wait_asynccnt 6" ::: "memory");
#endif
}
__device__ __forceinline__ void wait_async_0() {
#if __has_builtin(__builtin_amdgcn_s_wait_asynccnt)
  __builtin_amdgcn_s_wait_asynccnt(0);
#else
  asm volatile("s_wait_asynccnt 0" ::: "memory");
#endif
}

// pack 8 fp32 -> 8 bf16
__device__ __forceinline__ v8bf cvt8(float4 a, float4 b) {
  v8bf r;
  r[0] = (__bf16)a.x; r[1] = (__bf16)a.y; r[2] = (__bf16)a.z; r[3] = (__bf16)a.w;
  r[4] = (__bf16)b.x; r[5] = (__bf16)b.y; r[6] = (__bf16)b.z; r[7] = (__bf16)b.w;
  return r;
}

// ---------------------------------------------------------------------------
// Pack kernel: one block per 128x32 tile, stored in wave32 WMMA fragment
// order (ISA 7.12.2):
//   A-frag 16x32 bf16: lanes 0-15 hold K{0-7,16-23}; lanes 16-31 K{8-15,24-31}
//   B-frag 32x16 bf16: lane l: col n=l%16, K order same as A-frag
// Block id: [0, ABLK*KSTEPS)            -> Acat tile (rb, s)
//           [ABLK*KSTEPS, +WBLK*KSTEPS) -> Wcat tile (rb, s)
// ---------------------------------------------------------------------------
__global__ __launch_bounds__(256) void esn_pack(
    const float* __restrict__ passed,   // [BATCH x N_IN]
    const float* __restrict__ xin,      // [BATCH x N_RES]
    const float* __restrict__ lastOut,  // [N_OUT x BATCH]
    const float* __restrict__ lastPred, // [N_OUT x BATCH]
    const float* __restrict__ prob, const float* __restrict__ rnd,
    const float* __restrict__ W_in,     // [N_RES x N_IN]
    const float* __restrict__ W_res,    // [N_RES x N_RES]
    const float* __restrict__ W_out,    // [N_RES x N_OUT]
    __bf16* __restrict__ Acat, __bf16* __restrict__ Wcat) {

  const int blk = blockIdx.x;
  const bool isA = blk < ABLK * KSTEPS;
  const int rel = isA ? blk : blk - ABLK * KSTEPS;
  const int rb = rel / KSTEPS;
  const int s  = rel % KSTEPS;
  __bf16* dst = (isA ? Acat : Wcat) + ((size_t)rb * KSTEPS + s) * TILE_HALVES;

  const int tid = threadIdx.x;
  const int row = tid >> 1;          // 0..127
  const int halfk = tid & 1;         // which 16-wide half of the 32-k tile
  const int st = row >> 4, l0 = row & 15;
  const int off0 = (st * 32 + l0) * 16 + 8 * halfk;        // lanes 0-15 group
  const int off1 = (st * 32 + l0 + 16) * 16 + 8 * halfk;   // lanes 16-31 group

  const int k = s * 32 + halfk * 16;     // first of 16 consecutive K indices
  const int grow = rb * 128 + row;       // batch row (A) / reservoir row (W)

  float4 f0, f1, f2, f3;
  const float* rowp = nullptr;
  if (isA) {
    if (k < N_IN)              rowp = passed + (size_t)grow * N_IN + k;
    else if (k < N_IN + N_RES) rowp = xin + (size_t)grow * N_RES + (k - N_IN);
  } else {
    if (k < N_IN)              rowp = W_in  + (size_t)grow * N_IN  + k;
    else if (k < N_IN + N_RES) rowp = W_res + (size_t)grow * N_RES + (k - N_IN);
    else                       rowp = W_out + (size_t)grow * N_OUT + (k - N_IN - N_RES);
  }
  if (rowp) {
    const float4* p = (const float4*)rowp;
    f0 = p[0]; f1 = p[1]; f2 = p[2]; f3 = p[3];
  } else {
    // feedback region: A[b][k'] = (rand<prob ? lastOut : lastPred)[k'][b]
    const float* sel = (rnd[0] < prob[0]) ? lastOut : lastPred;
    const int kp = k - (N_IN + N_RES);
    float v[16];
#pragma unroll
    for (int i = 0; i < 16; ++i) v[i] = sel[(size_t)(kp + i) * BATCH + grow];
    f0 = make_float4(v[0], v[1], v[2], v[3]);
    f1 = make_float4(v[4], v[5], v[6], v[7]);
    f2 = make_float4(v[8], v[9], v[10], v[11]);
    f3 = make_float4(v[12], v[13], v[14], v[15]);
  }
  *(v8bf*)&dst[off0] = cvt8(f0, f1);
  *(v8bf*)&dst[off1] = cvt8(f2, f3);
}

// ---------------------------------------------------------------------------
// Fast-path GEMM.  Grid (16, 8); 256 threads = 8 wave32s in 2(M) x 4(N).
// Block tile 128(M) x 256(N); wave tile 64x64 = 4x4 WMMA tiles -> 16 wmma
// per 32-K step vs 16 ds_load_b128 of fragments (LDS-read balanced).
// ---------------------------------------------------------------------------
__global__ __launch_bounds__(256) void esn_wmma_async(
    const __bf16* __restrict__ Acat, const __bf16* __restrict__ Wcat,
    const float* __restrict__ x, const float* __restrict__ B_in,
    float* __restrict__ out) {

  __shared__ __align__(16) __bf16 lds[2][BUF_HALVES];  // 48KB of 320KB

  const int tid  = threadIdx.x;
  const int lane = tid & 31;
  const int wave = tid >> 5;
  const int wr   = wave >> 2;   // 0..1 (M)
  const int wc   = wave & 3;    // 0..3 (N)
  const int rbA  = blockIdx.y;  // 0..7   (128-row A block)
  const int nb   = blockIdx.x;  // 0..15  (256-col N block = two W tiles)

  const __bf16* gA  = Acat + (size_t)rbA * KSTEPS * TILE_HALVES;
  const __bf16* gB0 = Wcat + (size_t)(2 * nb)     * KSTEPS * TILE_HALVES;
  const __bf16* gB1 = Wcat + (size_t)(2 * nb + 1) * KSTEPS * TILE_HALVES;
  const uint32_t ldsBase = (uint32_t)(uintptr_t)&lds[0][0];

  v8f acc[4][4];
#pragma unroll
  for (int i = 0; i < 4; ++i)
#pragma unroll
    for (int j = 0; j < 4; ++j) acc[i][j] = (v8f)(0.0f);

  auto issue = [&](int s, int buf) {
    const size_t so = (size_t)s * TILE_HALVES + (size_t)tid * 16;
    const uint32_t lb = ldsBase + (uint32_t)buf * 24576u + (uint32_t)tid * 32u;
    async_b128(lb,           gA  + so);
    async_b128(lb + 16u,     gA  + so + 8);
    async_b128(lb + 8192u,   gB0 + so);
    async_b128(lb + 8208u,   gB0 + so + 8);
    async_b128(lb + 16384u,  gB1 + so);
    async_b128(lb + 16400u,  gB1 + so + 8);
  };

  issue(0, 0);
  for (int s = 0; s < KSTEPS; ++s) {
    const int cur = s & 1;
    if (s + 1 < KSTEPS) {        // prefetch next tile pair into other buffer
      issue(s + 1, cur ^ 1);
      wait_async_le6();          // step s's 6 copies done (in-order retire)
    } else {
      wait_async_0();
    }
    __syncthreads();             // all waves' tile-s data visible in LDS

    const __bf16* bufp = &lds[cur][0];
    v16bf af[4];
#pragma unroll
    for (int i = 0; i < 4; ++i)
      af[i] = *(const v16bf*)&bufp[((wr * 4 + i) * 32 + lane) * 16];
#pragma unroll
    for (int j = 0; j < 4; ++j) {
      const v16bf bf =
          *(const v16bf*)&bufp[4096 + ((wc * 4 + j) * 32 + lane) * 16];
#pragma unroll
      for (int i = 0; i < 4; ++i)
        acc[i][j] = __builtin_amdgcn_wmma_f32_16x16x32_bf16(
            false, af[i], false, bf, (short)0, acc[i][j], false, false);
    }
    __syncthreads();             // buffer `cur` free for step s+2's copies
  }

  // Epilogue: D layout: lane l<16 -> N=l, M=v ; lane>=16 -> N=l-16, M=v+8
  const int nn  = lane & 15;
  const int sel = (lane >> 4) * 8;
  const int mBase = rbA * 128, nBase = nb * 256;
#pragma unroll
  for (int j = 0; j < 4; ++j) {
    const int col = nBase + (wc * 4 + j) * 16 + nn;
    const float bin = B_in[col];
#pragma unroll
    for (int i = 0; i < 4; ++i) {
      const int rowBase = mBase + (wr * 4 + i) * 16 + sel;
#pragma unroll
      for (int v = 0; v < 8; ++v) {
        const size_t idx = (size_t)(rowBase + v) * N_RES + col;
        float pre = acc[i][j][v] + bin;
        float e  = __expf(2.0f * pre);                      // tanh via exp
        float th = 1.0f - 2.0f * __builtin_amdgcn_rcpf(e + 1.0f);
        out[idx] = 0.1f * x[idx] + 0.9f * th;
      }
    }
  }
}

// ===========================================================================
// Fallback path (small workspace): inline-convert GEMM
// ===========================================================================
__global__ __launch_bounds__(256) void esn_feedback_transpose(
    const float* __restrict__ lastOut, const float* __restrict__ lastPred,
    const float* __restrict__ prob, const float* __restrict__ rnd,
    float* __restrict__ fbT) {
  int idx = blockIdx.x * blockDim.x + threadIdx.x;
  const float* src = (rnd[0] < prob[0]) ? lastOut : lastPred;
  int b = idx >> 6, k = idx & 63;
  fbT[idx] = src[k * BATCH + b];
}

__device__ __forceinline__ void gemm_phase(
    const float* __restrict__ A, const float* __restrict__ W, int K,
    int mBase, int nBase, int tid, int lane, int wr, int wc,
    __bf16* ldsA, __bf16* ldsB, v8f (&acc)[4][2]) {
  const int row = tid >> 1, halfk = tid & 1;
  const int st = row >> 4, l0 = row & 15;
  const int off0 = (st * 32 + l0) * 16 + 8 * halfk;
  const int off1 = (st * 32 + l0 + 16) * 16 + 8 * halfk;
  const float* aRow = A + (size_t)(mBase + row) * K + halfk * 16;
  const float* wRow = W + (size_t)(nBase + row) * K + halfk * 16;
  for (int k0 = 0; k0 < K; k0 += 32) {
    const float4* ap = (const float4*)(aRow + k0);
    float4 a0 = ap[0], a1 = ap[1], a2 = ap[2], a3 = ap[3];
    const float4* wp = (const float4*)(wRow + k0);
    float4 b0 = wp[0], b1 = wp[1], b2 = wp[2], b3 = wp[3];
    *(v8bf*)&ldsA[off0] = cvt8(a0, a1);
    *(v8bf*)&ldsA[off1] = cvt8(a2, a3);
    *(v8bf*)&ldsB[off0] = cvt8(b0, b1);
    *(v8bf*)&ldsB[off1] = cvt8(b2, b3);
    __syncthreads();
    v16bf af[4], bfv[2];
#pragma unroll
    for (int i = 0; i < 4; ++i)
      af[i] = *(const v16bf*)&ldsA[((wr * 4 + i) * 32 + lane) * 16];
#pragma unroll
    for (int j = 0; j < 2; ++j)
      bfv[j] = *(const v16bf*)&ldsB[((wc * 2 + j) * 32 + lane) * 16];
#pragma unroll
    for (int i = 0; i < 4; ++i)
#pragma unroll
      for (int j = 0; j < 2; ++j)
        acc[i][j] = __builtin_amdgcn_wmma_f32_16x16x32_bf16(
            false, af[i], false, bfv[j], (short)0, acc[i][j], false, false);
    __syncthreads();
  }
}

__global__ __launch_bounds__(256) void esn_fused_step(
    const float* __restrict__ passed, const float* __restrict__ x,
    const float* __restrict__ fbT, const float* __restrict__ W_in,
    const float* __restrict__ W_res, const float* __restrict__ W_out,
    const float* __restrict__ B_in, float* __restrict__ out) {
  __shared__ __align__(16) __bf16 ldsA[TILE_HALVES];
  __shared__ __align__(16) __bf16 ldsB[TILE_HALVES];
  const int tid = threadIdx.x, lane = tid & 31, wave = tid >> 5;
  const int wr = wave >> 2, wc = wave & 3;
  const int mBase = blockIdx.y * 128, nBase = blockIdx.x * 128;
  v8f acc[4][2];
#pragma unroll
  for (int i = 0; i < 4; ++i)
#pragma unroll
    for (int j = 0; j < 2; ++j) acc[i][j] = (v8f)(0.0f);
  gemm_phase(passed, W_in,  N_IN,  mBase, nBase, tid, lane, wr, wc, ldsA, ldsB, acc);
  gemm_phase(x,      W_res, N_RES, mBase, nBase, tid, lane, wr, wc, ldsA, ldsB, acc);
  gemm_phase(fbT,    W_out, N_OUT, mBase, nBase, tid, lane, wr, wc, ldsA, ldsB, acc);
  const int nn = lane & 15, sel = (lane >> 4) * 8;
#pragma unroll
  for (int j = 0; j < 2; ++j) {
    const int col = nBase + (wc * 2 + j) * 16 + nn;
    const float bin = B_in[col];
#pragma unroll
    for (int i = 0; i < 4; ++i) {
      const int rowBase = mBase + (wr * 4 + i) * 16 + sel;
#pragma unroll
      for (int v = 0; v < 8; ++v) {
        const size_t idx = (size_t)(rowBase + v) * N_RES + col;
        float pre = acc[i][j][v] + bin;
        float e  = __expf(2.0f * pre);
        float th = 1.0f - 2.0f * __builtin_amdgcn_rcpf(e + 1.0f);
        out[idx] = 0.1f * x[idx] + 0.9f * th;
      }
    }
  }
}

// ---------------------------------------------------------------------------
extern "C" void kernel_launch(void* const* d_in, const int* in_sizes, int n_in,
                              void* d_out, int out_size, void* d_ws, size_t ws_size,
                              hipStream_t stream) {
  (void)in_sizes; (void)n_in; (void)out_size;
  const float* passed   = (const float*)d_in[0];
  const float* lastOut  = (const float*)d_in[1];
  const float* lastPred = (const float*)d_in[2];
  const float* x        = (const float*)d_in[3];
  const float* prob     = (const float*)d_in[4];
  const float* rnd      = (const float*)d_in[5];
  const float* W_in     = (const float*)d_in[6];
  const float* W_res    = (const float*)d_in[7];
  const float* W_out    = (const float*)d_in[8];
  const float* B_in     = (const float*)d_in[9];
  float* out = (float*)d_out;

  const size_t needA = (size_t)ABLK * KSTEPS * TILE_HALVES * sizeof(__bf16);
  const size_t needW = (size_t)WBLK * KSTEPS * TILE_HALVES * sizeof(__bf16);

  if (ws_size >= needA + needW) {
    __bf16* Acat = (__bf16*)d_ws;
    __bf16* Wcat = (__bf16*)((char*)d_ws + needA);
    esn_pack<<<(ABLK + WBLK) * KSTEPS, 256, 0, stream>>>(
        passed, x, lastOut, lastPred, prob, rnd, W_in, W_res, W_out, Acat, Wcat);
    dim3 grid(N_RES / 256, BATCH / 128);   // 16 x 8
    esn_wmma_async<<<grid, 256, 0, stream>>>(Acat, Wcat, x, B_in, out);
  } else {
    float* fbT = (float*)d_ws;   // 256 KB
    esn_feedback_transpose<<<(BATCH * N_OUT) / 256, 256, 0, stream>>>(
        lastOut, lastPred, prob, rnd, fbT);
    dim3 grid(N_RES / 128, BATCH / 128);
    esn_fused_step<<<grid, 256, 0, stream>>>(
        passed, x, fbT, W_in, W_res, W_out, B_in, out);
  }
}
